// HeteroGNN_42760694399652
// MI455X (gfx1250) — matrix operand order repaired and checked by reading, not compiled, not run on hardware
//
#include <hip/hip_runtime.h>
#include <hip/hip_bf16.h>

typedef __attribute__((ext_vector_type(16))) __bf16 v16bf;
typedef __attribute__((ext_vector_type(8)))  float  v8f;

#define TILE_M 128
#define TILE_N 128
#define TILE_K 32

static inline int cdiv(long a, long b) { return (int)((a + b - 1) / b); }

// ---------------- device helpers ----------------

// float atomic max via ordered-int trick (init buffer to very negative float)
__device__ inline void atomicMaxFloat(float* addr, float v) {
  int vi = __float_as_int(v);
  if (vi >= 0) atomicMax((int*)addr, vi);
  else         atomicMin((unsigned int*)addr, (unsigned int)vi);
}

// async 16B copy global -> LDS (no VGPR round-trip, tracked by ASYNCcnt)
__device__ inline void async_load_b128(unsigned lds_off, const void* gaddr) {
  asm volatile("global_load_async_to_lds_b128 %0, %1, off"
               :: "v"(lds_off), "v"(gaddr) : "memory");
}
__device__ inline void wait_asynccnt0() {
  asm volatile("s_wait_asynccnt 0x0" ::: "memory");
}

// ---------------- kernels ----------------

__global__ void fill_kernel(float* __restrict__ p, float v, int n) {
  int i = blockIdx.x * blockDim.x + threadIdx.x;
  if (i < n) p[i] = v;
}

// Bt[n*K + k] = bf16(B[k*N + n])  : one-shot transpose+convert of a weight matrix
__global__ void convert_transpose_kernel(const float* __restrict__ B, __bf16* __restrict__ Bt,
                                         int K, int N)
{
  int i = blockIdx.x * blockDim.x + threadIdx.x;
  if (i >= N * K) return;
  int n = i / K, k = i - n * K;
  Bt[i] = (__bf16)B[(size_t)k * N + n];
}

// C[M,N] = A[M,K] @ B[K,N] (+bias,+relu,+column sum/sumsq atomics).
// A: fp32 row-major. Bt: bf16 [N,K] row-major (pre-transposed weights).
// K % 32 == 0, N % 128 == 0. Block: 256 thr = 8 waves; wave w -> rows [w*16,w*16+16) x 128 cols.
// LDS tiles staged in WMMA fragment order: lane operand = 32 contiguous, 32B-aligned bytes.
// B tile is bf16 in global already -> staged with global_load_async_to_lds_b128 (ASYNCcnt path).
__global__ void __launch_bounds__(256)
gemm_bf16_kernel(const float* __restrict__ A, const __bf16* __restrict__ Bt,
                 const float* __restrict__ bias, float* __restrict__ C,
                 int M, int K, int N, int relu,
                 float* __restrict__ colsum, float* __restrict__ colsumsq)
{
  // Af[w][lane][16], Bf[j][lane][16]  (bf16)
  __shared__ __align__(32) __bf16 Af[8 * 32 * 16];
  __shared__ __align__(32) __bf16 Bf[8 * 32 * 16];

  const int tid  = threadIdx.x;
  const int lane = tid & 31;
  const int wave = tid >> 5;
  const int rowBase = blockIdx.x * TILE_M;
  const int colBase = blockIdx.y * TILE_N;

  v8f acc[8];
#pragma unroll
  for (int j = 0; j < 8; ++j)
#pragma unroll
    for (int e = 0; e < 8; ++e) acc[j][e] = 0.0f;

  for (int k0 = 0; k0 < K; k0 += TILE_K) {
    // ---- stage B fragments via async global->LDS copies (pure bf16 moves) ----
    // chunk c = (j*32 + l)*2 + half ; B-frag: f[0..7] <-> K=kb.. , f[8..15] <-> K=16+kb..
    for (int c = tid; c < 512; c += 256) {
      int half = c & 1;
      int jl   = c >> 1;
      int l    = jl & 31;
      int j    = jl >> 5;
      int n    = colBase + j * 16 + (l & 15);
      int kb   = (l >> 4) << 3;                  // 0 or 8
      int ks   = k0 + kb + half * 16;
      unsigned lds_off = (unsigned)(uintptr_t)(Bf + c * 8);
      async_load_b128(lds_off, Bt + (size_t)n * K + ks);
    }
    // ---- stage A fragments: fp32 -> bf16 convert through VGPRs ----
    // chunk c = (w*32 + l)*2 + half ; A-frag: f[0..7] <-> K=kb.. , f[8..15] <-> K=16+kb..
    for (int c = tid; c < 512; c += 256) {
      int half = c & 1;
      int wl   = c >> 1;
      int l    = wl & 31;
      int w    = wl >> 5;
      int gm   = rowBase + w * 16 + (l & 15);
      int kb   = (l >> 4) << 3;                  // 0 or 8
      int ks   = k0 + kb + half * 16;
      float fv[8];
      if (gm < M) {
        const float* p = A + (size_t)gm * K + ks;
        float4 a0 = *(const float4*)p;
        float4 a1 = *(const float4*)(p + 4);
        fv[0] = a0.x; fv[1] = a0.y; fv[2] = a0.z; fv[3] = a0.w;
        fv[4] = a1.x; fv[5] = a1.y; fv[6] = a1.z; fv[7] = a1.w;
      } else {
#pragma unroll
        for (int q = 0; q < 8; ++q) fv[q] = 0.0f;
      }
      union { __bf16 h[8]; uint4 u; } t;
#pragma unroll
      for (int q = 0; q < 8; ++q) t.h[q] = (__bf16)fv[q];
      *(uint4*)(Af + c * 8) = t.u;
    }
    wait_asynccnt0();          // async B chunks landed in LDS
    __syncthreads();           // all waves' staging visible

    v16bf af = *(const v16bf*)(Af + (wave * 32 + lane) * 16);
#pragma unroll
    for (int j = 0; j < 8; ++j) {
      v16bf bf = *(const v16bf*)(Bf + (j * 32 + lane) * 16);
      acc[j] = __builtin_amdgcn_wmma_f32_16x16x32_bf16(
          false, af, false, bf, (short)0, acc[j], false, false);
    }
    __syncthreads();
  }

  // epilogue: C/D layout -> (m = r + 8*(lane>=16), n = lane&15)
  const int mOff = rowBase + wave * 16 + ((lane >> 4) << 3);
#pragma unroll
  for (int j = 0; j < 8; ++j) {
    int gn = colBase + j * 16 + (lane & 15);
    float s = 0.0f, sq = 0.0f;
#pragma unroll
    for (int r = 0; r < 8; ++r) {
      int gm = mOff + r;
      if (gm < M) {
        float v = acc[j][r];
        if (bias) v += bias[gn];
        if (relu) v = fmaxf(v, 0.0f);
        C[(size_t)gm * N + gn] = v;
        s += v; sq += v * v;
      }
    }
    if (colsum) { atomicAdd(&colsum[gn], s); atomicAdd(&colsumsq[gn], sq); }
  }
}

// y = gamma*(h-mu)*rsqrt(var+eps)+beta, biased var from column sums
__global__ void bn_apply_kernel(const float* __restrict__ h, const float* __restrict__ sum,
                                const float* __restrict__ ssq, const float* __restrict__ gamma,
                                const float* __restrict__ beta, int M, float* __restrict__ out)
{
  int i = blockIdx.x * blockDim.x + threadIdx.x;
  if (i >= M * 128) return;
  int c = i & 127;
  float invM  = 1.0f / (float)M;
  float mu    = sum[c] * invM;
  float var   = ssq[c] * invM - mu * mu;
  float scale = gamma[c] * rsqrtf(var + 1e-5f);
  out[i] = (h[i] - mu) * scale + beta[c];
}

// wvec[k*4+h] = sum_c W[k, h*128+c] * a[h,c]   (W: [128,512] row-major, a: [4,128])
__global__ void head_weight_kernel(const float* __restrict__ W, const float* __restrict__ a,
                                   float* __restrict__ wvec)
{
  int t = threadIdx.x;            // 512 threads
  int k = t >> 2, h = t & 3;
  float s = 0.0f;
#pragma unroll 4
  for (int c = 0; c < 128; ++c)
    s += W[k * 512 + h * 128 + c] * a[h * 128 + c];
  wvec[k * 4 + h] = s;
}

// l[i,h] = sum_k enc[i,k] * wvec[k,h]
__global__ void node_logit_kernel(const float* __restrict__ x, const float* __restrict__ wvec,
                                  float* __restrict__ out, int N)
{
  int t = blockIdx.x * blockDim.x + threadIdx.x;
  if (t >= N * 4) return;
  int i = t >> 2, h = t & 3;
  const float* row = x + (size_t)i * 128;
  float s = 0.0f;
#pragma unroll 8
  for (int k = 0; k < 128; ++k) s += row[k] * wvec[k * 4 + h];
  out[t] = s;
}

__device__ inline void decode_edge(const int* esrc, const int* edst, int E, int i,
                                   int& src, int& dst, bool& ok) {
  if (i < E) { src = esrc[i]; dst = edst[i]; ok = (src != dst); }
  else       { src = dst = i - E; ok = true; }
}

__global__ void edge_max_kernel(const int* __restrict__ esrc, const int* __restrict__ edst,
                                int E, int nloop,
                                const float* __restrict__ ls, const float* __restrict__ ld,
                                float* __restrict__ m)
{
  int i = blockIdx.x * blockDim.x + threadIdx.x;
  if (i >= E + nloop) return;
  int src, dst; bool ok;
  decode_edge(esrc, edst, E, i, src, dst, ok);
#pragma unroll
  for (int h = 0; h < 4; ++h) {
    float x  = ls[src * 4 + h] + ld[dst * 4 + h];
    float lg = x > 0.0f ? x : 0.2f * x;           // leaky_relu(0.2)
    atomicMaxFloat(&m[dst * 4 + h], ok ? lg : -1.0e30f);
  }
}

__global__ void edge_denom_kernel(const int* __restrict__ esrc, const int* __restrict__ edst,
                                  int E, int nloop,
                                  const float* __restrict__ ls, const float* __restrict__ ld,
                                  const float* __restrict__ m, float* __restrict__ den)
{
  int i = blockIdx.x * blockDim.x + threadIdx.x;
  if (i >= E + nloop) return;
  int src, dst; bool ok;
  decode_edge(esrc, edst, E, i, src, dst, ok);
  if (!ok) return;
#pragma unroll
  for (int h = 0; h < 4; ++h) {
    float x  = ls[src * 4 + h] + ld[dst * 4 + h];
    float lg = x > 0.0f ? x : 0.2f * x;
    atomicAdd(&den[dst * 4 + h], __expf(lg - m[dst * 4 + h]));
  }
}

// one wave per edge: lanes coalesce over the 128 channels, 4 heads folded per lane
__global__ void __launch_bounds__(256)
edge_aggr_kernel(const int* __restrict__ esrc, const int* __restrict__ edst, int E, int nloop,
                 const float* __restrict__ ls, const float* __restrict__ ld,
                 const float* __restrict__ m, const float* __restrict__ den,
                 const float* __restrict__ xs, float* __restrict__ out)
{
  int wave = threadIdx.x >> 5, lane = threadIdx.x & 31;
  int e = blockIdx.x * 8 + wave;
  if (e >= E + nloop) return;
  int src, dst; bool ok;
  decode_edge(esrc, edst, E, e, src, dst, ok);
  if (!ok) return;                                // masked edge: alpha == 0
  float alpha[4];
#pragma unroll
  for (int h = 0; h < 4; ++h) {
    float x  = ls[src * 4 + h] + ld[dst * 4 + h];
    float lg = x > 0.0f ? x : 0.2f * x;
    float p  = __expf(lg - m[dst * 4 + h]);
    float d  = den[dst * 4 + h];
    alpha[h] = p / (d > 0.0f ? d : 1.0f);
  }
  const float* xrow = xs + (size_t)src * 512;
#pragma unroll
  for (int rep = 0; rep < 4; ++rep) {
    int c = lane + rep * 32;
    float v = alpha[0] * xrow[c]       + alpha[1] * xrow[128 + c]
            + alpha[2] * xrow[256 + c] + alpha[3] * xrow[384 + c];
    atomicAdd(&out[(size_t)dst * 128 + c], v);
  }
}

// feat += mean_over_heads(out) + bias   (0.25 = 1/H applied here)
__global__ void final_add_kernel(float* __restrict__ feat, const float* __restrict__ out,
                                 const float* __restrict__ bias, int Ndst)
{
  int i = blockIdx.x * blockDim.x + threadIdx.x;
  if (i >= Ndst * 128) return;
  feat[i] += out[i] * 0.25f + bias[i & 127];
}

// ---------------- host orchestration ----------------

static void run_relation(const float* src_feat, int Nsrc,
                         float* dst_feat, int Ndst,
                         const int* esrc, const int* edst, int E, int nloop,
                         const float* Ws, const float* Wd,
                         const float* a_s, const float* a_d, const float* bias,
                         float* xs, float* outb, float* ls, float* ld,
                         float* mbuf, float* den, float* wvec, __bf16* bt,
                         hipStream_t stream)
{
  int Etot = E + nloop;
  head_weight_kernel<<<1, 512, 0, stream>>>(Ws, a_s, wvec);
  node_logit_kernel<<<cdiv((long)Nsrc * 4, 256), 256, 0, stream>>>(src_feat, wvec, ls, Nsrc);
  head_weight_kernel<<<1, 512, 0, stream>>>(Wd, a_d, wvec);
  node_logit_kernel<<<cdiv((long)Ndst * 4, 256), 256, 0, stream>>>(dst_feat, wvec, ld, Ndst);
  convert_transpose_kernel<<<cdiv(128L * 512, 256), 256, 0, stream>>>(Ws, bt, 128, 512);
  gemm_bf16_kernel<<<dim3(cdiv(Nsrc, 128), 4), 256, 0, stream>>>(
      src_feat, bt, nullptr, xs, Nsrc, 128, 512, 0, nullptr, nullptr);
  fill_kernel<<<cdiv((long)Ndst * 4, 256), 256, 0, stream>>>(mbuf, -3.0e38f, Ndst * 4);
  fill_kernel<<<cdiv((long)Ndst * 4, 256), 256, 0, stream>>>(den, 0.0f, Ndst * 4);
  fill_kernel<<<cdiv((long)Ndst * 128, 256), 256, 0, stream>>>(outb, 0.0f, Ndst * 128);
  edge_max_kernel  <<<cdiv(Etot, 256), 256, 0, stream>>>(esrc, edst, E, nloop, ls, ld, mbuf);
  edge_denom_kernel<<<cdiv(Etot, 256), 256, 0, stream>>>(esrc, edst, E, nloop, ls, ld, mbuf, den);
  edge_aggr_kernel <<<cdiv(Etot, 8),   256, 0, stream>>>(esrc, edst, E, nloop, ls, ld, mbuf, den, xs, outb);
  final_add_kernel <<<cdiv((long)Ndst * 128, 256), 256, 0, stream>>>(dst_feat, outb, bias, Ndst);
}

extern "C" void kernel_launch(void* const* d_in, const int* in_sizes, int n_in,
                              void* d_out, int out_size, void* d_ws, size_t ws_size,
                              hipStream_t stream)
{
  (void)n_in; (void)out_size; (void)ws_size;
  const float* x_gene = (const float*)d_in[0];
  const float* x_dis  = (const float*)d_in[1];
  const int*   e1s = (const int*)d_in[2];
  const int*   e1d = (const int*)d_in[3];
  const int*   e2s = (const int*)d_in[4];
  const int*   e2d = (const int*)d_in[5];
  const float* Wg = (const float*)d_in[6];
  const float* bg = (const float*)d_in[7];
  const float* gg = (const float*)d_in[8];
  const float* betag = (const float*)d_in[9];
  const float* Wd = (const float*)d_in[10];
  const float* bd = (const float*)d_in[11];
  const float* gd = (const float*)d_in[12];
  const float* betad = (const float*)d_in[13];
  const float* W1s = (const float*)d_in[14];
  const float* W1d = (const float*)d_in[15];
  const float* a1s = (const float*)d_in[16];
  const float* a1d = (const float*)d_in[17];
  const float* b1  = (const float*)d_in[18];
  const float* W2s = (const float*)d_in[19];
  const float* W2d = (const float*)d_in[20];
  const float* a2s = (const float*)d_in[21];
  const float* a2d = (const float*)d_in[22];
  const float* b2  = (const float*)d_in[23];

  const int Ngene = in_sizes[0] / 512;
  const int Ndis  = in_sizes[1] / 256;
  const int E1 = in_sizes[2];
  const int E2 = in_sizes[4];
  const int nloop1 = (Ndis < Ngene) ? Ndis : Ngene;

  float* gene = (float*)d_out;                           // [Ngene,128]
  float* dis  = (float*)d_out + (size_t)Ngene * 128;     // [Ndis,128]

  // workspace layout (floats)
  float* ws = (float*)d_ws;
  float* xs   = ws;                                       // max Ngene*512
  float* hbuf = xs   + (size_t)Ngene * 512;               // reused as GAT out-accum (Ngene*128)
  float* ls   = hbuf + (size_t)Ngene * 128;               // Ngene*4
  float* ld   = ls   + (size_t)Ngene * 4;                 // Ngene*4
  float* mbuf = ld   + (size_t)Ngene * 4;                 // Ngene*4
  float* den  = mbuf + (size_t)Ngene * 4;                 // Ngene*4
  float* sumb = den  + (size_t)Ngene * 4;                 // 128
  float* ssqb = sumb + 128;                               // 128
  float* wvec = ssqb + 128;                               // 512
  __bf16* bt  = (__bf16*)(wvec + 512);                    // 65536 bf16 (= 32768 floats)

  // ---- encode gene: relu(x@Wg+bg) -> BN -> gene ----
  fill_kernel<<<1, 256, 0, stream>>>(sumb, 0.0f, 128);
  fill_kernel<<<1, 256, 0, stream>>>(ssqb, 0.0f, 128);
  convert_transpose_kernel<<<cdiv(128L * 512, 256), 256, 0, stream>>>(Wg, bt, 512, 128);
  gemm_bf16_kernel<<<dim3(cdiv(Ngene, 128), 1), 256, 0, stream>>>(
      x_gene, bt, bg, hbuf, Ngene, 512, 128, 1, sumb, ssqb);
  bn_apply_kernel<<<cdiv((long)Ngene * 128, 256), 256, 0, stream>>>(
      hbuf, sumb, ssqb, gg, betag, Ngene, gene);

  // ---- encode disease ----
  fill_kernel<<<1, 256, 0, stream>>>(sumb, 0.0f, 128);
  fill_kernel<<<1, 256, 0, stream>>>(ssqb, 0.0f, 128);
  convert_transpose_kernel<<<cdiv(128L * 256, 256), 256, 0, stream>>>(Wd, bt, 256, 128);
  gemm_bf16_kernel<<<dim3(cdiv(Ndis, 128), 1), 256, 0, stream>>>(
      x_dis, bt, bd, hbuf, Ndis, 256, 128, 1, sumb, ssqb);
  bn_apply_kernel<<<cdiv((long)Ndis * 128, 256), 256, 0, stream>>>(
      hbuf, sumb, ssqb, gd, betad, Ndis, dis);

  // ---- relation 1: disease -> gene ----
  run_relation(dis, Ndis, gene, Ngene, e1s, e1d, E1, nloop1,
               W1s, W1d, a1s, a1d, b1, xs, hbuf, ls, ld, mbuf, den, wvec, bt, stream);

  // ---- relation 2: gene -> gene (updated features) ----
  run_relation(gene, Ngene, gene, Ngene, e2s, e2d, E2, Ngene,
               W2s, W2d, a2s, a2d, b2, xs, hbuf, ls, ld, mbuf, den, wvec, bt, stream);
}
